// TransformerBlock_11940009083300
// MI455X (gfx1250) — matrix operand order repaired and checked by reading, not compiled, not run on hardware
//
#include <hip/hip_runtime.h>

typedef __attribute__((ext_vector_type(16))) __bf16 v16bf;
typedef __attribute__((ext_vector_type(8)))  __bf16 v8bf;
typedef __attribute__((ext_vector_type(8)))  float  v8f;

#define NCH 64
#define TILE_ROWS 16
#define WAVES 8
#define ROWS_PER_BLOCK (WAVES * TILE_ROWS)   // 128

// ---------------------------------------------------------------------------
// Fragment helpers (wave32, 16x16x32 bf16 WMMA)
// A (16x32 bf16): lane m = lane&15, half = lane>>4.
//   slots 0..7  -> k = kchunk*32 + half*8 + s        (contiguous 16B)
//   slots 8..15 -> k = kchunk*32 + 16 + half*8 + s-8 (contiguous 16B)
// B (32x16 bf16): lane n = lane&15; slot s -> same k as A slot s (half*8+s / 16+half*8+s-8)
//   -> 16 contiguous bf16 per lane when pre-packed.
// C/D (16x16 f32): element r -> m = r + 8*(lane>>4), n = lane&15.
// ---------------------------------------------------------------------------
__device__ inline v16bf make_a_frag(const __bf16* row) {
  v8bf lo = *(const v8bf*)(row);
  v8bf hi = *(const v8bf*)(row + 16);
  v16bf a;
#pragma unroll
  for (int i = 0; i < 8; ++i) { a[i] = lo[i]; a[i + 8] = hi[i]; }
  return a;
}

// One 16x64 @ 64x64 GEMM (+bias, optional relu): 8 WMMAs.
// tilerow = &lds_tile[m0*64]; pmat = packed weight matrix (4096 bf16).
__device__ inline void gemm64(const __bf16* tilerow, int half, int lane,
                              const __bf16* __restrict__ pmat,
                              const float* __restrict__ bias, bool relu,
                              v8f* acc) {
  v16bf a0 = make_a_frag(tilerow + 0 * 32 + half * 8);
  v16bf a1 = make_a_frag(tilerow + 1 * 32 + half * 8);
  const int n0 = lane & 15;
#pragma unroll
  for (int t = 0; t < 4; ++t) {
    float bv = bias ? bias[t * 16 + n0] : 0.0f;
    v8f c;
#pragma unroll
    for (int r = 0; r < 8; ++r) c[r] = bv;
    v16bf b0 = *(const v16bf*)(pmat + ((0 * 4 + t) * 32 + lane) * 16);
    c = __builtin_amdgcn_wmma_f32_16x16x32_bf16(false, a0, false, b0,
                                                (short)0, c, false, false);
    v16bf b1 = *(const v16bf*)(pmat + ((1 * 4 + t) * 32 + lane) * 16);
    c = __builtin_amdgcn_wmma_f32_16x16x32_bf16(false, a1, false, b1,
                                                (short)0, c, false, false);
    if (relu) {
#pragma unroll
      for (int r = 0; r < 8; ++r) c[r] = fmaxf(c[r], 0.0f);
    }
    acc[t] = c;
  }
}

// ---------------------------------------------------------------------------
// Pack 64x64 row-major f32 weights -> bf16 B-fragment layout.
// mats: 0 W_in, 1 W_src, 2 W_dst, 3 W_lin, 4 W_out, 5 pos_w2, 6 att_w1, 7 att_w2
// ---------------------------------------------------------------------------
__global__ void __launch_bounds__(256) pack_weights_kernel(
    const float* __restrict__ W_in,  const float* __restrict__ W_src,
    const float* __restrict__ W_dst, const float* __restrict__ W_lin,
    const float* __restrict__ W_out, const float* __restrict__ pos_w2,
    const float* __restrict__ att_w1, const float* __restrict__ att_w2,
    __bf16* __restrict__ packed) {
  const float* mats[8] = {W_in, W_src, W_dst, W_lin, W_out, pos_w2, att_w1, att_w2};
  for (int i = blockIdx.x * blockDim.x + threadIdx.x; i < 8 * 4096;
       i += gridDim.x * blockDim.x) {
    int mat = i >> 12;
    int idx = i & 4095;
    int s = idx & 15;
    int L = (idx >> 4) & 31;
    int t = (idx >> 9) & 3;
    int f = idx >> 11;
    int k = f * 32 + (s < 8 ? (L >> 4) * 8 + s : 16 + (L >> 4) * 8 + (s - 8));
    int n = t * 16 + (L & 15);
    packed[i] = (__bf16)mats[mat][k * NCH + n];
  }
}

// ---------------------------------------------------------------------------
// Node input kernel: h = relu(x@W_in+b_in); as=h@W_src, ad=h@W_dst, v=h@W_lin
// ---------------------------------------------------------------------------
__global__ void __launch_bounds__(256) node_in_kernel(
    const float* __restrict__ x, const float* __restrict__ b_in,
    const __bf16* __restrict__ packed,
    __bf16* __restrict__ as_bf, __bf16* __restrict__ ad_bf,
    __bf16* __restrict__ v_bf, int N) {
  __shared__ alignas(16) __bf16 s_tile[WAVES][TILE_ROWS * NCH];
  const int tid = threadIdx.x, w = tid >> 5, lane = tid & 31;
  const int m0 = lane & 15, half = lane >> 4;
  const long base = (long)blockIdx.x * ROWS_PER_BLOCK + w * TILE_ROWS;
  long row = base + m0;
  long rowc = row < N ? row : (long)N - 1;

#pragma unroll
  for (int c0 = 0; c0 < 32; ++c0) {
    int c = half * 32 + c0;
    s_tile[w][m0 * NCH + c] = (__bf16)x[rowc * NCH + c];
  }
  __syncthreads();

  v8f h[4];
  gemm64(&s_tile[w][m0 * NCH], half, lane, packed + 0 * 4096, b_in, true, h);
  __syncthreads();
#pragma unroll
  for (int t = 0; t < 4; ++t)
#pragma unroll
    for (int r = 0; r < 8; ++r)
      s_tile[w][(r + 8 * half) * NCH + t * 16 + m0] = (__bf16)h[t][r];
  __syncthreads();

  __bf16* outs[3] = {as_bf, ad_bf, v_bf};
  const int mat[3] = {1, 2, 3};
#pragma unroll
  for (int o = 0; o < 3; ++o) {
    v8f acc[4];
    gemm64(&s_tile[w][m0 * NCH], half, lane, packed + mat[o] * 4096, nullptr,
           false, acc);
#pragma unroll
    for (int t = 0; t < 4; ++t)
#pragma unroll
      for (int r = 0; r < 8; ++r) {
        long rr = base + r + 8 * half;
        if (rr < N) outs[o][rr * NCH + t * 16 + m0] = (__bf16)acc[t][r];
      }
  }
}

// ---------------------------------------------------------------------------
// Edge kernel (templated on MODE). 16 edges per wave, 128 per block.
//  MODE 0: a = attn MLP; atomic int-max into amax (a >= 0 after relu).
//  MODE 1: recompute a; ex=exp(a-amax); atomic add denom, accum numerator.
// ---------------------------------------------------------------------------
template <int MODE>
__global__ void __launch_bounds__(256) edge_kernel(
    const int* __restrict__ esrc, const int* __restrict__ edst,
    const float* __restrict__ pos,
    const float* __restrict__ pw1, const float* __restrict__ pb1,
    const float* __restrict__ pb2, const float* __restrict__ ab1,
    const float* __restrict__ ab2,
    const __bf16* __restrict__ packed,
    const __bf16* __restrict__ as_bf, const __bf16* __restrict__ ad_bf,
    const __bf16* __restrict__ v_bf,
    int* __restrict__ amax, float* __restrict__ denom,
    float* __restrict__ accum, int E) {
  __shared__ alignas(16) __bf16 s_tile[WAVES][TILE_ROWS * NCH];
  __shared__ int s_s[WAVES][16], s_d[WAVES][16], s_vld[WAVES][16];
  __shared__ float s_dx[WAVES][16], s_dy[WAVES][16], s_dz[WAVES][16];

  const int tid = threadIdx.x, w = tid >> 5, lane = tid & 31;
  const int m0 = lane & 15, half = lane >> 4;
  const long base = (long)blockIdx.x * ROWS_PER_BLOCK + w * TILE_ROWS;

  if (half == 0) {
    long e = base + m0;
    int vld = e < E;
    long ec = vld ? e : (long)E - 1;
    int s = esrc[ec], d = edst[ec];
    s_s[w][m0] = s; s_d[w][m0] = d; s_vld[w][m0] = vld;
    s_dx[w][m0] = pos[(long)d * 3 + 0] - pos[(long)s * 3 + 0];
    s_dy[w][m0] = pos[(long)d * 3 + 1] - pos[(long)s * 3 + 1];
    s_dz[w][m0] = pos[(long)d * 3 + 2] - pos[(long)s * 3 + 2];
  }
  __syncthreads();

  // pos_nn layer 1 (K=3) on VALU: hid1 = relu(dp @ pos_w1 + pos_b1)
  {
    float dx = s_dx[w][m0], dy = s_dy[w][m0], dz = s_dz[w][m0];
#pragma unroll
    for (int c0 = 0; c0 < 32; ++c0) {
      int c = half * 32 + c0;
      float h = fmaf(dx, pw1[c],
                 fmaf(dy, pw1[64 + c], fmaf(dz, pw1[128 + c], pb1[c])));
      s_tile[w][m0 * NCH + c] = (__bf16)fmaxf(h, 0.0f);
    }
  }
  __syncthreads();

  // delta = relu(hid1 @ pos_w2 + pos_b2)  -> kept in C fragments
  v8f dacc[4];
  gemm64(&s_tile[w][m0 * NCH], half, lane, packed + 5 * 4096, pb2, true, dacc);
  __syncthreads();

  // attn input = alpha_dst[d] - alpha_src[s] + delta (gathers hit L2)
#pragma unroll
  for (int t = 0; t < 4; ++t)
#pragma unroll
    for (int r = 0; r < 8; ++r) {
      int c = t * 16 + m0, m = r + 8 * half;
      long d = s_d[w][m], s = s_s[w][m];
      float vv = (float)ad_bf[d * NCH + c] - (float)as_bf[s * NCH + c] +
                 dacc[t][r];
      s_tile[w][m * NCH + c] = (__bf16)vv;
    }
  __syncthreads();

  v8f h1[4];
  gemm64(&s_tile[w][m0 * NCH], half, lane, packed + 6 * 4096, ab1, true, h1);
  __syncthreads();
#pragma unroll
  for (int t = 0; t < 4; ++t)
#pragma unroll
    for (int r = 0; r < 8; ++r)
      s_tile[w][(r + 8 * half) * NCH + t * 16 + m0] = (__bf16)h1[t][r];
  __syncthreads();

  v8f av[4];
  gemm64(&s_tile[w][m0 * NCH], half, lane, packed + 7 * 4096, ab2, true, av);

#pragma unroll
  for (int t = 0; t < 4; ++t)
#pragma unroll
    for (int r = 0; r < 8; ++r) {
      int c = t * 16 + m0, m = r + 8 * half;
      if (!s_vld[w][m]) continue;
      long d = s_d[w][m];
      if (MODE == 0) {
        // a >= 0 (relu) -> int bit-pattern max == float max
        atomicMax(&amax[d * NCH + c], __float_as_int(av[t][r]));
      } else {
        long s = s_s[w][m];
        float am = __int_as_float(amax[d * NCH + c]);
        float ex = __expf(av[t][r] - am);
        atomicAdd(&denom[d * NCH + c], ex);
        float vv = (float)v_bf[s * NCH + c] + dacc[t][r];
        atomicAdd(&accum[d * NCH + c], ex * vv);
      }
    }
}

// ---------------------------------------------------------------------------
// Output kernel: out = relu((accum / (denom + 1e-16)) @ W_out + b_out)
// ---------------------------------------------------------------------------
__global__ void __launch_bounds__(256) node_out_kernel(
    const float* __restrict__ accum, const float* __restrict__ denom,
    const __bf16* __restrict__ packed, const float* __restrict__ b_out,
    float* __restrict__ out, int N) {
  __shared__ alignas(16) __bf16 s_tile[WAVES][TILE_ROWS * NCH];
  const int tid = threadIdx.x, w = tid >> 5, lane = tid & 31;
  const int m0 = lane & 15, half = lane >> 4;
  const long base = (long)blockIdx.x * ROWS_PER_BLOCK + w * TILE_ROWS;
  long row = base + m0;
  long rowc = row < N ? row : (long)N - 1;

#pragma unroll
  for (int c0 = 0; c0 < 32; ++c0) {
    int c = half * 32 + c0;
    long i = rowc * NCH + c;
    float v = accum[i] / (denom[i] + 1e-16f);
    s_tile[w][m0 * NCH + c] = (__bf16)v;
  }
  __syncthreads();

  v8f acc[4];
  gemm64(&s_tile[w][m0 * NCH], half, lane, packed + 4 * 4096, b_out, true, acc);
#pragma unroll
  for (int t = 0; t < 4; ++t)
#pragma unroll
    for (int r = 0; r < 8; ++r) {
      long rr = base + r + 8 * half;
      if (rr < N) out[rr * NCH + t * 16 + m0] = fmaxf(acc[t][r], 0.0f);
    }
}

// ---------------------------------------------------------------------------
extern "C" void kernel_launch(void* const* d_in, const int* in_sizes, int n_in,
                              void* d_out, int out_size, void* d_ws,
                              size_t ws_size, hipStream_t stream) {
  const float* x      = (const float*)d_in[0];
  const float* pos    = (const float*)d_in[1];
  const int*   eidx   = (const int*)d_in[2];
  const float* W_in   = (const float*)d_in[3];
  const float* b_in   = (const float*)d_in[4];
  const float* W_out  = (const float*)d_in[5];
  const float* b_out  = (const float*)d_in[6];
  const float* W_lin  = (const float*)d_in[7];
  const float* W_src  = (const float*)d_in[8];
  const float* W_dst  = (const float*)d_in[9];
  const float* pos_w1 = (const float*)d_in[10];
  const float* pos_b1 = (const float*)d_in[11];
  const float* pos_w2 = (const float*)d_in[12];
  const float* pos_b2 = (const float*)d_in[13];
  const float* att_w1 = (const float*)d_in[14];
  const float* att_b1 = (const float*)d_in[15];
  const float* att_w2 = (const float*)d_in[16];
  const float* att_b2 = (const float*)d_in[17];

  const int N = in_sizes[0] / NCH;
  const int E = in_sizes[2] / 2;
  const int* esrc = eidx;
  const int* edst = eidx + E;

  // Workspace layout (all region sizes are multiples of 256B)
  char* ws = (char*)d_ws;
  size_t off = 0;
  auto alloc = [&](size_t bytes) -> void* {
    void* p = (void*)(ws + off);
    off += (bytes + 255) & ~(size_t)255;
    return p;
  };
  __bf16* packed = (__bf16*)alloc((size_t)8 * 4096 * sizeof(__bf16));
  __bf16* as_bf  = (__bf16*)alloc((size_t)N * NCH * sizeof(__bf16));
  __bf16* ad_bf  = (__bf16*)alloc((size_t)N * NCH * sizeof(__bf16));
  __bf16* v_bf   = (__bf16*)alloc((size_t)N * NCH * sizeof(__bf16));
  int*    amax   = (int*)alloc((size_t)N * NCH * sizeof(int));
  float*  denom  = (float*)alloc((size_t)N * NCH * sizeof(float));
  float*  accum  = (float*)alloc((size_t)N * NCH * sizeof(float));

  hipMemsetAsync(amax,  0, (size_t)N * NCH * sizeof(int),   stream);
  hipMemsetAsync(denom, 0, (size_t)N * NCH * sizeof(float), stream);
  hipMemsetAsync(accum, 0, (size_t)N * NCH * sizeof(float), stream);

  pack_weights_kernel<<<32, 256, 0, stream>>>(W_in, W_src, W_dst, W_lin, W_out,
                                              pos_w2, att_w1, att_w2, packed);

  int node_blocks = (N + ROWS_PER_BLOCK - 1) / ROWS_PER_BLOCK;
  node_in_kernel<<<node_blocks, 256, 0, stream>>>(x, b_in, packed, as_bf,
                                                  ad_bf, v_bf, N);

  int edge_blocks = (E + ROWS_PER_BLOCK - 1) / ROWS_PER_BLOCK;
  edge_kernel<0><<<edge_blocks, 256, 0, stream>>>(
      esrc, edst, pos, pos_w1, pos_b1, pos_b2, att_b1, att_b2, packed, as_bf,
      ad_bf, v_bf, amax, denom, accum, E);
  edge_kernel<1><<<edge_blocks, 256, 0, stream>>>(
      esrc, edst, pos, pos_w1, pos_b1, pos_b2, att_b1, att_b2, packed, as_bf,
      ad_bf, v_bf, amax, denom, accum, E);

  node_out_kernel<<<node_blocks, 256, 0, stream>>>(accum, denom, packed, b_out,
                                                   (float*)d_out, N);
}
